// HANsBlock_68143951118557
// MI455X (gfx1250) — matrix operand order repaired and checked by reading, not compiled, not run on hardware
//
#include <hip/hip_runtime.h>
#include <hip/hip_bf16.h>
#include <math.h>

typedef __attribute__((ext_vector_type(16))) __bf16 v16bf;
typedef __attribute__((ext_vector_type(4)))  __bf16 v4bf;
typedef __attribute__((ext_vector_type(8)))  float  v8f;
// 4-byte-aligned float4 for rows whose byte offset is only DWORD-aligned (K=166)
typedef __attribute__((ext_vector_type(4), aligned(4))) float f4u;

// ---------------------------------------------------------------------------
// WMMA fragment loaders (16x16x32 bf16, wave32 layouts per CDNA5 ISA 7.12.2)
// A: 16x32 (MxK). lanes 0-15: M=lane, K = {0..7,16..23}; lanes 16-31: K={8..15,24..31}
// B: 32x16 (KxN), sourced from W[N,K] row-major. lanes 0-15: K={0..7,16..23}; 16-31: K={8..15,24..31}
// ---------------------------------------------------------------------------
__device__ __forceinline__ v16bf load_a_frag(const __bf16* s, int ld, int lane) {
    const int m  = lane & 15;
    const int kb = (lane < 16) ? 0 : 8;
    const __bf16* p = s + m * ld;
    v16bf a;
#pragma unroll
    for (int j = 0; j < 8; ++j) a[j] = p[kb + j];
#pragma unroll
    for (int j = 0; j < 8; ++j) a[8 + j] = p[kb + 16 + j];
    return a;
}

__device__ __forceinline__ v16bf load_b_frag(const __bf16* s, int ld, int lane) {
    const int n  = lane & 15;
    const int kb = (lane < 16) ? 0 : 8;
    const __bf16* p = s + n * ld;
    v16bf b;
#pragma unroll
    for (int j = 0; j < 8; ++j) b[j] = p[kb + j];
#pragma unroll
    for (int j = 0; j < 8; ++j) b[8 + j] = p[kb + 16 + j];
    return b;
}

__device__ __forceinline__ v4bf cvt4(f4u v) {
    v4bf b;
    b[0] = (__bf16)v.x; b[1] = (__bf16)v.y;
    b[2] = (__bf16)v.z; b[3] = (__bf16)v.w;
    return b;
}

__device__ __forceinline__ f4u relu4f(f4u v) {
    v.x = fmaxf(v.x, 0.0f); v.y = fmaxf(v.y, 0.0f);
    v.z = fmaxf(v.z, 0.0f); v.w = fmaxf(v.w, 0.0f);
    return v;
}

// ---------------------------------------------------------------------------
// Stage one 32(k) step of X (32 rows) and W (128 rows) into LDS as bf16.
// Fast path: issue ALL five b128 loads first (distinct locals -> distinct
// VGPRs -> loads pipelined, single s_wait_loadcnt), then cvt + ds_store_b64.
// ---------------------------------------------------------------------------
template <bool RELU_IN>
__device__ __forceinline__ void stage_tiles(const float* __restrict__ X,
                                            const float* __restrict__ W,
                                            __bf16* sA, __bf16* sW, int LDT,
                                            int rowBase, int N, int K, int k0,
                                            int tid, bool fullM)
{
    if (fullM && (k0 + 32 <= K)) {
        const int r  = tid >> 3;
        const int c4 = (tid & 7) << 2;
        const float* xp = X + (size_t)(rowBase + r) * K + (k0 + c4);
        const float* wp = W + (size_t)r * K + (k0 + c4);
        f4u va  = *(const f4u*)xp;
        f4u vw0 = *(const f4u*)(wp);
        f4u vw1 = *(const f4u*)(wp + (size_t)32 * K);
        f4u vw2 = *(const f4u*)(wp + (size_t)64 * K);
        f4u vw3 = *(const f4u*)(wp + (size_t)96 * K);
        if (RELU_IN) va = relu4f(va);
        *(v4bf*)(sA + r * LDT + c4)        = cvt4(va);
        *(v4bf*)(sW + (r +  0) * LDT + c4) = cvt4(vw0);
        *(v4bf*)(sW + (r + 32) * LDT + c4) = cvt4(vw1);
        *(v4bf*)(sW + (r + 64) * LDT + c4) = cvt4(vw2);
        *(v4bf*)(sW + (r + 96) * LDT + c4) = cvt4(vw3);
    } else {
        // rare ragged tile (last k-step of K=166, or ragged row tile)
        for (int i = tid; i < 32 * 32; i += 256) {
            int r = i >> 5, c = i & 31;
            int gr = rowBase + r, gc = k0 + c;
            float v = (gr < N && gc < K) ? X[(size_t)gr * K + gc] : 0.0f;
            if (RELU_IN) v = fmaxf(v, 0.0f);
            sA[r * LDT + c] = (__bf16)v;
        }
        for (int i = tid; i < 128 * 32; i += 256) {
            int r = i >> 5, c = i & 31;
            int gc = k0 + c;
            float v = (gc < K) ? W[(size_t)r * K + gc] : 0.0f;
            sW[r * LDT + c] = (__bf16)v;
        }
    }
}

// ---------------------------------------------------------------------------
// Double-buffered bf16 WMMA core. Block tile 32x128; 8 waves = 2 row-strips
// x 4 col-strips; 2 WMMA tiles/wave/k-step. sA/sW are 2x-sized buffers.
// ---------------------------------------------------------------------------
template <bool RELU_IN>
__device__ __forceinline__ void gemm_core(const float* __restrict__ X,
                                          const float* __restrict__ W,
                                          __bf16* sA, __bf16* sW, int LDT,
                                          v8f& acc0, v8f& acc1,
                                          int rowBase, int N, int K,
                                          int tid, int lane, int waveRow, int waveCol)
{
    const bool fullM = (rowBase + 32 <= N);
    const int  aBuf  = 32 * LDT;
    const int  wBuf  = 128 * LDT;

    stage_tiles<RELU_IN>(X, W, sA, sW, LDT, rowBase, N, K, 0, tid, fullM);
    __syncthreads();

    int cur = 0;
    for (int k0 = 0; k0 < K; k0 += 32) {
        const int nxt = cur ^ 1;
        if (k0 + 32 < K)   // stage next tile while WMMAs consume current one
            stage_tiles<RELU_IN>(X, W, sA + nxt * aBuf, sW + nxt * wBuf, LDT,
                                 rowBase, N, K, k0 + 32, tid, fullM);

        v16bf af = load_a_frag(sA + cur * aBuf + waveRow * 16 * LDT, LDT, lane);
        v16bf b0 = load_b_frag(sW + cur * wBuf + (waveCol * 32 + 0)  * LDT, LDT, lane);
        v16bf b1 = load_b_frag(sW + cur * wBuf + (waveCol * 32 + 16) * LDT, LDT, lane);
        acc0 = __builtin_amdgcn_wmma_f32_16x16x32_bf16(false, af, false, b0,
                                                       (short)0, acc0, false, false);
        acc1 = __builtin_amdgcn_wmma_f32_16x16x32_bf16(false, af, false, b1,
                                                       (short)0, acc1, false, false);
        __syncthreads();
        cur = nxt;
    }
}

__launch_bounds__(256, 2)
__global__ void gemm_bias_act_kernel(const float* __restrict__ X,
                                     const float* __restrict__ W,
                                     const float* __restrict__ bias,
                                     float* __restrict__ Y,
                                     int N, int K, int act)
{
    constexpr int LDT = 40;                  // bf16 elems; keeps v4bf stores 8B aligned
    __shared__ __align__(16) __bf16 sA[2 * 32 * LDT];
    __shared__ __align__(16) __bf16 sW[2 * 128 * LDT];

    const int tid     = threadIdx.x;
    const int lane    = tid & 31;
    const int wid     = tid >> 5;
    const int waveRow = wid >> 2;            // 0..1  (16 rows each)
    const int waveCol = wid & 3;             // 0..3  (32 cols each)
    const int rowBase = blockIdx.x * 32;

    v8f acc0 = {};
    v8f acc1 = {};
    gemm_core<false>(X, W, sA, sW, LDT, acc0, acc1, rowBase, N, K,
                     tid, lane, waveRow, waveCol);

    const int nn  = lane & 15;
    const int mLo = (lane < 16) ? 0 : 8;
#pragma unroll
    for (int r = 0; r < 8; ++r) {
        int gm = rowBase + waveRow * 16 + mLo + r;
        if (gm < N) {
            int c0 = waveCol * 32 + nn;
            float v0 = acc0[r] + bias[c0];
            float v1 = acc1[r] + bias[c0 + 16];
            if (act) { v0 = fmaxf(v0, 0.0f); v1 = fmaxf(v1, 0.0f); }
            Y[(size_t)gm * 128 + c0]      = v0;
            Y[(size_t)gm * 128 + c0 + 16] = v1;
        }
    }
}

// ---------------------------------------------------------------------------
// colsum[n] += sum_rows tanh(relu(X) @ Wk^T + bk)[.,n]  (semantic attention)
// relu fused into the bf16 staging; per-block LDS column reduce (ds_add_f32),
// then 128 global atomics per block.
// ---------------------------------------------------------------------------
__launch_bounds__(256, 2)
__global__ void tanh_colsum_kernel(const float* __restrict__ X,
                                   const float* __restrict__ Wk,
                                   const float* __restrict__ bk,
                                   float* __restrict__ colsum,
                                   int N)
{
    constexpr int LDT = 40;
    __shared__ __align__(16) __bf16 sA[2 * 32 * LDT];
    __shared__ __align__(16) __bf16 sW[2 * 128 * LDT];
    __shared__ float colAcc[128];

    const int tid     = threadIdx.x;
    const int lane    = tid & 31;
    const int wid     = tid >> 5;
    const int waveRow = wid >> 2;
    const int waveCol = wid & 3;
    const int rowBase = blockIdx.x * 32;

    if (tid < 128) colAcc[tid] = 0.0f;

    v8f acc0 = {};
    v8f acc1 = {};
    gemm_core<true>(X, Wk, sA, sW, LDT, acc0, acc1, rowBase, N, 128,
                    tid, lane, waveRow, waveCol);

    const int nn  = lane & 15;
    const int mLo = (lane < 16) ? 0 : 8;
#pragma unroll
    for (int r = 0; r < 8; ++r) {
        int gm = rowBase + waveRow * 16 + mLo + r;
        if (gm < N) {
            int c0 = waveCol * 32 + nn;
            atomicAdd(&colAcc[c0],      tanhf(acc0[r] + bk[c0]));
            atomicAdd(&colAcc[c0 + 16], tanhf(acc1[r] + bk[c0 + 16]));
        }
    }
    __syncthreads();
    if (tid < 128) atomicAdd(&colsum[tid], colAcc[tid]);
}

// ---------------------------------------------------------------------------
// Ordered-uint encoding so float max can use integer atomicMax.
// ---------------------------------------------------------------------------
__device__ __forceinline__ unsigned fenc(float f) {
    unsigned u = __float_as_uint(f);
    return (u & 0x80000000u) ? ~u : (u | 0x80000000u);
}
__device__ __forceinline__ float fdec(unsigned u) {
    u = (u & 0x80000000u) ? (u & 0x7FFFFFFFu) : ~u;
    return __uint_as_float(u);
}

// a[node,h] = <h[node, h*16 : h*16+16], att[h]>
__global__ void att_score_kernel(const float* __restrict__ h,
                                 const float* __restrict__ att,
                                 float* __restrict__ out, int Nn)
{
    int idx = blockIdx.x * blockDim.x + threadIdx.x;
    if (idx >= Nn * 8) return;
    int node = idx >> 3, hh = idx & 7;
    const float* hp = h + (size_t)node * 128 + hh * 16;
    const float* ap = att + hh * 16;
    float s = 0.0f;
#pragma unroll
    for (int j = 0; j < 16; ++j) s += hp[j] * ap[j];
    out[idx] = s;
}

// pass 1: alpha = leakyrelu(a_s[row] + a_d[col], 0.2); per-(dst,head) max
__global__ void edge_alpha_max_kernel(const int* __restrict__ row,
                                      const int* __restrict__ col,
                                      const float* __restrict__ a_s,
                                      const float* __restrict__ a_d,
                                      float* __restrict__ alpha,
                                      unsigned* __restrict__ mx, int E)
{
    int idx = blockIdx.x * blockDim.x + threadIdx.x;
    if (idx >= E * 8) return;
    int e = idx >> 3, h = idx & 7;
    float a = a_s[(size_t)row[e] * 8 + h] + a_d[(size_t)col[e] * 8 + h];
    a = (a > 0.0f) ? a : 0.2f * a;
    alpha[idx] = a;
    atomicMax(&mx[(size_t)col[e] * 8 + h], fenc(a));
}

// pass 2: alpha = exp(alpha - max); per-(dst,head) sum
__global__ void edge_exp_sum_kernel(const int* __restrict__ col,
                                    float* __restrict__ alpha,
                                    const unsigned* __restrict__ mx,
                                    float* __restrict__ ssum, int E)
{
    int idx = blockIdx.x * blockDim.x + threadIdx.x;
    if (idx >= E * 8) return;
    int e = idx >> 3, h = idx & 7;
    size_t d = (size_t)col[e] * 8 + h;
    float v = __expf(alpha[idx] - fdec(mx[d]));
    alpha[idx] = v;
    atomicAdd(&ssum[d], v);
}

// pass 3: one wave per edge; lane owns 4 contiguous features (b128 load)
__global__ void edge_scatter_kernel(const int* __restrict__ row,
                                    const int* __restrict__ col,
                                    const float* __restrict__ hsrc,
                                    const float* __restrict__ alpha,
                                    const float* __restrict__ ssum,
                                    float* __restrict__ out, int E)
{
    int gw   = (blockIdx.x * blockDim.x + threadIdx.x) >> 5;
    int lane = threadIdx.x & 31;
    if (gw >= E) return;
    int r = row[gw], c = col[gw];
    int h = lane >> 2;                                     // head 0..7
    float w = alpha[(size_t)gw * 8 + h] /
              (ssum[(size_t)c * 8 + h] + 1e-16f);
    float4 v = *((const float4*)(hsrc + (size_t)r * 128) + lane);
    float* dst = out + (size_t)c * 128 + lane * 4;
    atomicAdd(dst + 0, v.x * w);
    atomicAdd(dst + 1, v.y * w);
    atomicAdd(dst + 2, v.z * w);
    atomicAdd(dst + 3, v.w * w);
}

__global__ void fill_u32_kernel(unsigned* __restrict__ p, unsigned v, size_t n) {
    size_t i = (size_t)blockIdx.x * blockDim.x + threadIdx.x;
    if (i < n) p[i] = v;
}

// semantic softmax over 2 metapath scores; 128 threads
__global__ void beta_kernel(const float* __restrict__ colsum,
                            const float* __restrict__ q,
                            float* __restrict__ beta, float invN)
{
    __shared__ float red[128];
    int t = threadIdx.x;
    float qv = q[t];
    red[t] = qv * colsum[t] * invN;
    __syncthreads();
    for (int s = 64; s > 0; s >>= 1) { if (t < s) red[t] += red[t + s]; __syncthreads(); }
    float s0 = red[0];
    __syncthreads();
    red[t] = qv * colsum[128 + t] * invN;
    __syncthreads();
    for (int s = 64; s > 0; s >>= 1) { if (t < s) red[t] += red[t + s]; __syncthreads(); }
    if (t == 0) {
        float s1 = red[0];
        float m  = fmaxf(s0, s1);
        float e0 = __expf(s0 - m), e1 = __expf(s1 - m);
        float inv = 1.0f / (e0 + e1);
        beta[0] = e0 * inv;
        beta[1] = e1 * inv;
    }
}

// out = relu(beta0 * relu(a1) + beta1 * relu(out))   (out_a2 lives in d_out;
// relu of the segment-sum results is fused here instead of separate passes)
__global__ void combine_kernel(const float* __restrict__ a1,
                               const float* __restrict__ beta,
                               float* out, size_t n4)
{
    size_t i = (size_t)blockIdx.x * blockDim.x + threadIdx.x;
    if (i >= n4) return;
    float b0 = beta[0], b1 = beta[1];
    float4 x = ((const float4*)a1)[i];
    float4 y = ((float4*)out)[i];
    x.x = fmaxf(x.x, 0.0f); x.y = fmaxf(x.y, 0.0f);
    x.z = fmaxf(x.z, 0.0f); x.w = fmaxf(x.w, 0.0f);
    y.x = fmaxf(y.x, 0.0f); y.y = fmaxf(y.y, 0.0f);
    y.z = fmaxf(y.z, 0.0f); y.w = fmaxf(y.w, 0.0f);
    float4 r;
    r.x = fmaxf(b0 * x.x + b1 * y.x, 0.0f);
    r.y = fmaxf(b0 * x.y + b1 * y.y, 0.0f);
    r.z = fmaxf(b0 * x.z + b1 * y.z, 0.0f);
    r.w = fmaxf(b0 * x.w + b1 * y.w, 0.0f);
    ((float4*)out)[i] = r;
}

// ---------------------------------------------------------------------------
extern "C" void kernel_launch(void* const* d_in, const int* in_sizes, int n_in,
                              void* d_out, int out_size, void* d_ws, size_t ws_size,
                              hipStream_t stream)
{
    const float* x_addr = (const float*)d_in[0];
    const float* x_tx   = (const float*)d_in[1];
    // d_in[2] (edge_st), d_in[13], d_in[14] are dead code in the reference.
    const int*   e_ts   = (const int*)d_in[3];
    const int*   e_aa   = (const int*)d_in[4];
    const float* Wa  = (const float*)d_in[5];
    const float* ba  = (const float*)d_in[6];
    const float* Wt  = (const float*)d_in[7];
    const float* bt  = (const float*)d_in[8];
    const float* Pa  = (const float*)d_in[9];
    const float* pba = (const float*)d_in[10];
    const float* Pt  = (const float*)d_in[11];
    const float* pbt = (const float*)d_in[12];
    const float* as_ts = (const float*)d_in[15];
    const float* ad_ts = (const float*)d_in[16];
    const float* as_aa = (const float*)d_in[17];
    const float* ad_aa = (const float*)d_in[18];
    const float* Wk  = (const float*)d_in[19];
    const float* bk  = (const float*)d_in[20];
    const float* q   = (const float*)d_in[21];

    const int FA = in_sizes[5] / 128;     // 166
    const int FT = in_sizes[7] / 128;     // 64
    const int NA = in_sizes[0] / FA;      // 100000
    const int NT = in_sizes[1] / FT;      // 100000
    const int E  = in_sizes[3] / 2;       // 1000000
    const int NMAX = (NA > NT) ? NA : NT;

    // ---- workspace carving (aliased through the pipeline) ----
    char* wsp = (char*)d_ws;
    size_t off = 0;
    auto carve = [&](size_t bytes) -> char* {
        char* p = wsp + off;
        off += (bytes + 255) & ~(size_t)255;
        return p;
    };
    float*    ha     = (float*)carve((size_t)NA * 128 * 4);
    float*    ht     = (float*)carve((size_t)NT * 128 * 4);
    float*    bufA   = (float*)carve((size_t)NA * 128 * 4);   // xa, then out_a1
    size_t    bufBsz = (size_t)NT * 128 * 4;
    size_t    alphsz = (size_t)E * 8 * 4;
    float*    bufB   = (float*)carve(bufBsz > alphsz ? bufBsz : alphsz); // xt, then alpha
    float*    aS     = (float*)carve((size_t)NMAX * 8 * 4);
    float*    aD     = (float*)carve((size_t)NA * 8 * 4);
    unsigned* mx     = (unsigned*)carve((size_t)NA * 8 * 4);
    float*    ssum   = (float*)carve((size_t)NA * 8 * 4);
    float*    colsum = (float*)carve(2 * 128 * 4);
    float*    beta   = (float*)carve(2 * 4);
    (void)ws_size; (void)n_in; (void)out_size;

    float* outA1 = bufA;
    float* outA2 = (float*)d_out;
    float* alpha = bufB;

    dim3 blk(256);
    auto gblocks = [](size_t n) { return dim3((unsigned)((n + 255) / 256)); };
    auto mblocks = [](int N)    { return dim3((unsigned)((N + 31) / 32)); };

    // 1) input projections (relu) + per-type HAN projections
    gemm_bias_act_kernel<<<mblocks(NA), blk, 0, stream>>>(x_addr, Wa, ba, bufA, NA, FA, 1);
    gemm_bias_act_kernel<<<mblocks(NT), blk, 0, stream>>>(x_tx,   Wt, bt, bufB, NT, FT, 1);
    gemm_bias_act_kernel<<<mblocks(NA), blk, 0, stream>>>(bufA, Pa, pba, ha, NA, 128, 0);
    gemm_bias_act_kernel<<<mblocks(NT), blk, 0, stream>>>(bufB, Pt, pbt, ht, NT, 128, 0);

    // 2) per-edge-type GAT attention + message passing (dst type = address)
    auto run_edge = [&](const int* edges, const float* hsrc, int Nsrc,
                        const float* hdst, int Ndst,
                        const float* attS, const float* attD, float* outbuf) {
        const int* row = edges;
        const int* col = edges + E;
        att_score_kernel<<<gblocks((size_t)Nsrc * 8), blk, 0, stream>>>(hsrc, attS, aS, Nsrc);
        att_score_kernel<<<gblocks((size_t)Ndst * 8), blk, 0, stream>>>(hdst, attD, aD, Ndst);
        size_t n8   = (size_t)Ndst * 8;
        size_t nout = (size_t)Ndst * 128;
        fill_u32_kernel<<<gblocks(n8), blk, 0, stream>>>(mx, 0x00800000u, n8); // enc(-FLT_MAX)
        fill_u32_kernel<<<gblocks(n8), blk, 0, stream>>>((unsigned*)ssum, 0u, n8);
        fill_u32_kernel<<<gblocks(nout), blk, 0, stream>>>((unsigned*)outbuf, 0u, nout);
        size_t ne8 = (size_t)E * 8;
        edge_alpha_max_kernel<<<gblocks(ne8), blk, 0, stream>>>(row, col, aS, aD, alpha, mx, E);
        edge_exp_sum_kernel<<<gblocks(ne8), blk, 0, stream>>>(col, alpha, mx, ssum, E);
        edge_scatter_kernel<<<gblocks((size_t)E * 32), blk, 0, stream>>>(row, col, hsrc,
                                                                         alpha, ssum, outbuf, E);
    };

    run_edge(e_ts, ht, NT, ha, NA, as_ts, ad_ts, outA1);   // transaction -> address
    run_edge(e_aa, ha, NA, ha, NA, as_aa, ad_aa, outA2);   // address -> address

    // 3) semantic (metapath) attention + final combine (relu of outs fused in)
    fill_u32_kernel<<<dim3(1), blk, 0, stream>>>((unsigned*)colsum, 0u, 256);
    tanh_colsum_kernel<<<mblocks(NA), blk, 0, stream>>>(outA1, Wk, bk, colsum,       NA);
    tanh_colsum_kernel<<<mblocks(NA), blk, 0, stream>>>(outA2, Wk, bk, colsum + 128, NA);
    beta_kernel<<<dim3(1), dim3(128), 0, stream>>>(colsum, q, beta, 1.0f / (float)NA);

    size_t n4 = (size_t)NA * 128 / 4;
    combine_kernel<<<gblocks(n4), blk, 0, stream>>>(outA1, beta, (float*)d_out, n4);
}
